// Cnn_Medium_84344567759556
// MI455X (gfx1250) — 8-bit matrix operand order repaired and checked by reading, compiled, not run on hardware
//
#include <hip/hip_runtime.h>
#include <math.h>


typedef __attribute__((ext_vector_type(16))) int   v16i;
typedef __attribute__((ext_vector_type(8)))  float v8f;

#define FP8_P1 ((unsigned char)0x38)
#define FP8_M1 ((unsigned char)0xB8)

__device__ __forceinline__ unsigned char sign_fp8(float t) {
    return t > 0.f ? FP8_P1 : (t < 0.f ? FP8_M1 : (unsigned char)0x00);
}

__device__ __forceinline__ void async_copy16(void* lds_dst, const void* gsrc) {
    unsigned lds_off = (unsigned)(unsigned long long)lds_dst;
    asm volatile("global_load_async_to_lds_b128 %0, %1, off"
                 :: "v"(lds_off), "v"(gsrc) : "memory");
}
__device__ __forceinline__ void wait_async0() {
    asm volatile("s_wait_asynccnt 0" ::: "memory");
}

__device__ __forceinline__ v16i load_a_frag(const unsigned char* At, int lane) {
    const int row   = lane & 15;
    const int khalf = (lane >> 4) << 4;
    const unsigned char* r = At + row * 128;
    v16i a;
#pragma unroll
    for (int h = 0; h < 2; ++h) {
#pragma unroll
        for (int p = 0; p < 4; ++p) {
            const uint2 d = *(const uint2*)(r + h * 64 + (p >> 1) * 32 + (p & 1) * 8 + khalf);
            a[h * 8 + p * 2]     = (int)d.x;
            a[h * 8 + p * 2 + 1] = (int)d.y;
        }
    }
    return a;
}

__device__ __forceinline__ v16i load_b_frag(const unsigned char* Bt, int lane) {
    const int col  = lane & 15;
    const int koff = (lane >> 4) << 4;
    const unsigned char* cp = Bt + col * 128;
    v16i b;
#pragma unroll
    for (int q = 0; q < 4; ++q) {
        const uint4 d = *(const uint4*)(cp + q * 32 + koff);
        b[q * 4 + 0] = (int)d.x; b[q * 4 + 1] = (int)d.y;
        b[q * 4 + 2] = (int)d.z; b[q * 4 + 3] = (int)d.w;
    }
    return b;
}

__device__ __forceinline__ v8f wmma_fp8(v16i a, v16i b, v8f c) {
    return __builtin_amdgcn_wmma_f32_16x16x128_fp8_fp8(a, b, (short)0, c,
                                                       false, false);
}

template <int CIN, int COUT, int H, int W>
__global__ __launch_bounds__(256) void bconv_gemm(
    const unsigned char* __restrict__ act,
    const unsigned char* __restrict__ wgt,
    float* __restrict__ outf) {
    constexpr int K   = 9 * CIN;
    constexpr int NKC = K / 128;

    __shared__ unsigned char Atile[2][128 * 128];
    __shared__ unsigned char Btile[2][64 * 128];

    const int m0   = blockIdx.x * 128;
    const int n0   = blockIdx.y * 64;
    const int tid  = threadIdx.x;
    const int lane = tid & 31;
    const int wave = tid >> 5;
    const int wm   = (wave & 3) * 32;
    const int wn   = (wave >> 2) * 32;

    v8f c00 = {}, c01 = {}, c10 = {}, c11 = {};

    const int arow = tid >> 1;
    const int aseg = (tid & 1) * 64;
    const int am = m0 + arow;
    const int ab = am / (H * W);
    const int apx = am % (H * W);
    const int ay = apx / W, ax = apx % W;
    const int bcol = tid >> 2;
    const int bseg = (tid & 3) * 32;

    auto stage = [&](int kc, int buf) {
        const int kpos = (kc * 128) / CIN;
        const int cin0 = (kc * 128) % CIN;
        const int yy   = ay + kpos / 3 - 1;
        const int xx   = ax + kpos % 3 - 1;
        unsigned char* adst = &Atile[buf][arow * 128 + aseg];
        if (yy >= 0 && yy < H && xx >= 0 && xx < W) {
            const unsigned char* asrc =
                act + ((size_t)((ab * H + yy) * W + xx)) * CIN + cin0 + aseg;
            async_copy16(adst,      asrc);
            async_copy16(adst + 16, asrc + 16);
            async_copy16(adst + 32, asrc + 32);
            async_copy16(adst + 48, asrc + 48);
        } else {
            *(uint4*)(adst)      = uint4{0, 0, 0, 0};
            *(uint4*)(adst + 16) = uint4{0, 0, 0, 0};
            *(uint4*)(adst + 32) = uint4{0, 0, 0, 0};
            *(uint4*)(adst + 48) = uint4{0, 0, 0, 0};
        }
        const unsigned char* bsrc =
            wgt + (size_t)(n0 + bcol) * K + kc * 128 + bseg;
        unsigned char* bdst = &Btile[buf][bcol * 128 + bseg];
        async_copy16(bdst,      bsrc);
        async_copy16(bdst + 16, bsrc + 16);
    };

    stage(0, 0);
    for (int kc = 0; kc < NKC; ++kc) {
        const int buf = kc & 1;
        wait_async0();
        __syncthreads();
        if (kc + 1 < NKC) stage(kc + 1, buf ^ 1);

        v16i a0 = load_a_frag(&Atile[buf][(wm)*128],      lane);
        v16i a1 = load_a_frag(&Atile[buf][(wm + 16)*128], lane);
        v16i b0 = load_b_frag(&Btile[buf][(wn)*128],      lane);
        v16i b1 = load_b_frag(&Btile[buf][(wn + 16)*128], lane);
        c00 = wmma_fp8(a0, b0, c00);
        c01 = wmma_fp8(a0, b1, c01);
        c10 = wmma_fp8(a1, b0, c10);
        c11 = wmma_fp8(a1, b1, c11);
    }

    const int nn  = n0 + wn + (lane & 15);
    const int mb0 = m0 + wm + (lane >> 4) * 8;
#pragma unroll
    for (int v = 0; v < 8; ++v) {
        const size_t r0 = (size_t)(mb0 + v);
        const size_t r1 = (size_t)(mb0 + 16 + v);
        outf[r0 * COUT + nn]      = fmaxf(c00[v], 0.f);
        outf[r0 * COUT + nn + 16] = fmaxf(c01[v], 0.f);
        outf[r1 * COUT + nn]      = fmaxf(c10[v], 0.f);
        outf[r1 * COUT + nn + 16] = fmaxf(c11[v], 0.f);
    }
}

__global__ __launch_bounds__(256) void conv1_kernel(
    const float* __restrict__ x, const float* __restrict__ w1,
    const float* __restrict__ b1, const float* __restrict__ s1,
    const float* __restrict__ bb1, unsigned char* __restrict__ out8) {
    const size_t idx = (size_t)blockIdx.x * blockDim.x + threadIdx.x;
    const int c  = idx & 127;
    const size_t p = idx >> 7;
    const int xx = p % 32;
    const int yy = (p / 32) % 32;
    const int b  = (int)(p / 1024);
    float acc = 0.f;
#pragma unroll
    for (int ky = 0; ky < 3; ++ky) {
        const int y2 = yy + ky - 1;
        if (y2 < 0 || y2 >= 32) continue;
#pragma unroll
        for (int kx = 0; kx < 3; ++kx) {
            const int x2 = xx + kx - 1;
            if (x2 < 0 || x2 >= 32) continue;
            const float* xp = x + ((size_t)(b * 32 + y2) * 32 + x2) * 3;
            const float* wp = w1 + (size_t)((ky * 3 + kx) * 3) * 128 + c;
            acc = fmaf(xp[0], wp[0],   acc);
            acc = fmaf(xp[1], wp[128], acc);
            acc = fmaf(xp[2], wp[256], acc);
        }
    }
    float h = fmaxf(acc + b1[c], 0.f);
    out8[idx] = sign_fp8(h * s1[c] + bb1[c]);
}

__global__ void binarize_weights(const float* __restrict__ w,
                                 unsigned char* __restrict__ o, int K, int N) {
    const size_t idx = (size_t)blockIdx.x * blockDim.x + threadIdx.x;
    if (idx >= (size_t)K * N) return;
    const int k = (int)(idx % K);
    const int n = (int)(idx / K);
    o[idx] = sign_fp8(w[(size_t)k * N + n]);
}

template <bool POOL, bool BIN>
__global__ void postproc(const float* __restrict__ in, int H, int W, int C,
                         const float* __restrict__ scale,
                         const float* __restrict__ bias,
                         unsigned char* __restrict__ out8,
                         float* __restrict__ outf, size_t total) {
    const size_t idx = (size_t)blockIdx.x * blockDim.x + threadIdx.x;
    if (idx >= total) return;
    const int HO = POOL ? H / 2 : H, WO = POOL ? W / 2 : W;
    const int c = (int)(idx % C);
    size_t p = idx / C;
    const int xo = (int)(p % WO); p /= WO;
    const int yo = (int)(p % HO);
    const int b  = (int)(p / HO);
    float v;
    if (POOL) {
        const size_t base = ((size_t)(b * H + 2 * yo) * W + 2 * xo) * C + c;
        const size_t rs = (size_t)W * C;
        v = fmaxf(fmaxf(in[base], in[base + C]),
                  fmaxf(in[base + rs], in[base + rs + C]));
    } else {
        v = in[((size_t)(b * H + yo) * W + xo) * C + c];
    }
    const float t = v * scale[c] + bias[c];
    if (BIN) out8[idx] = sign_fp8(t);
    else     outf[idx] = t;
}

__global__ __launch_bounds__(256) void dense_softmax(
    const float* __restrict__ h, const float* __restrict__ dw,
    const float* __restrict__ db, float* __restrict__ out) {
    const int row  = blockIdx.x * 8 + (threadIdx.x >> 5);
    const int lane = threadIdx.x & 31;
    const float* hp = h + (size_t)row * 512;
    float acc[10];
#pragma unroll
    for (int j = 0; j < 10; ++j) acc[j] = 0.f;
    for (int c = lane; c < 512; c += 32) {
        const float hv = hp[c];
#pragma unroll
        for (int j = 0; j < 10; ++j) acc[j] = fmaf(hv, dw[c * 10 + j], acc[j]);
    }
#pragma unroll
    for (int j = 0; j < 10; ++j)
        for (int off = 16; off; off >>= 1) acc[j] += __shfl_xor(acc[j], off, 32);
    float mx = -1e30f;
#pragma unroll
    for (int j = 0; j < 10; ++j) { acc[j] += db[j]; mx = fmaxf(mx, acc[j]); }
    float se = 0.f;
#pragma unroll
    for (int j = 0; j < 10; ++j) { acc[j] = __expf(acc[j] - mx); se += acc[j]; }
    const float inv = 1.f / se;
    if (lane < 10) {
        float v = 0.f;
#pragma unroll
        for (int j = 0; j < 10; ++j) if (lane == j) v = acc[j];
        out[(size_t)row * 10 + lane] = v * inv;
    }
}

extern "C" void kernel_launch(void* const* d_in, const int* in_sizes, int n_in,
                              void* d_out, int out_size, void* d_ws,
                              size_t ws_size, hipStream_t stream) {
    (void)in_sizes; (void)n_in; (void)out_size; (void)ws_size;
    const float* x    = (const float*)d_in[0];
    const float* w1   = (const float*)d_in[1];
    const float* b1   = (const float*)d_in[2];
    const float* w2   = (const float*)d_in[3];
    const float* w3   = (const float*)d_in[4];
    const float* w4   = (const float*)d_in[5];
    const float* w5   = (const float*)d_in[6];
    const float* w6   = (const float*)d_in[7];
    const float* bn1s = (const float*)d_in[8];
    const float* bn1b = (const float*)d_in[9];
    const float* bn2s = (const float*)d_in[10];
    const float* bn2b = (const float*)d_in[11];
    const float* bn3s = (const float*)d_in[12];
    const float* bn3b = (const float*)d_in[13];
    const float* bn4s = (const float*)d_in[14];
    const float* bn4b = (const float*)d_in[15];
    const float* bn5s = (const float*)d_in[16];
    const float* bn5b = (const float*)d_in[17];
    const float* bn6s = (const float*)d_in[18];
    const float* bn6b = (const float*)d_in[19];
    const float* dw   = (const float*)d_in[20];
    const float* db   = (const float*)d_in[21];

    unsigned char* base = (unsigned char*)d_ws;
    size_t off = 0;
    auto carve = [&](size_t bytes) {
        unsigned char* p = base + off;
        off += (bytes + 255) & ~(size_t)255;
        return p;
    };
    unsigned char* wq2 = carve(1152u * 128);
    unsigned char* wq3 = carve(1152u * 256);
    unsigned char* wq4 = carve(2304u * 256);
    unsigned char* wq5 = carve(2304u * 512);
    unsigned char* wq6 = carve(4608u * 512);
    unsigned char* actA = carve((size_t)128 * 32 * 32 * 128);
    unsigned char* actB = carve((size_t)128 * 32 * 32 * 128);
    float* tmp = (float*)carve((size_t)128 * 32 * 32 * 128 * 4);

    const int T = 256;
    auto blk = [](size_t n) { return (unsigned)((n + 255) / 256); };

    binarize_weights<<<blk(1152u * 128), T, 0, stream>>>(w2, wq2, 1152, 128);
    binarize_weights<<<blk(1152u * 256), T, 0, stream>>>(w3, wq3, 1152, 256);
    binarize_weights<<<blk(2304u * 256), T, 0, stream>>>(w4, wq4, 2304, 256);
    binarize_weights<<<blk(2304u * 512), T, 0, stream>>>(w5, wq5, 2304, 512);
    binarize_weights<<<blk(4608u * 512), T, 0, stream>>>(w6, wq6, 4608, 512);

    conv1_kernel<<<blk((size_t)128 * 32 * 32 * 128), T, 0, stream>>>(
        x, w1, b1, bn1s, bn1b, actA);

    bconv_gemm<128, 128, 32, 32><<<dim3(131072 / 128, 128 / 64), T, 0, stream>>>(
        actA, wq2, tmp);
    postproc<true, true><<<blk((size_t)128 * 16 * 16 * 128), T, 0, stream>>>(
        tmp, 32, 32, 128, bn2s, bn2b, actB, nullptr, (size_t)128 * 16 * 16 * 128);

    bconv_gemm<128, 256, 16, 16><<<dim3(32768 / 128, 256 / 64), T, 0, stream>>>(
        actB, wq3, tmp);
    postproc<false, true><<<blk((size_t)128 * 16 * 16 * 256), T, 0, stream>>>(
        tmp, 16, 16, 256, bn3s, bn3b, actA, nullptr, (size_t)128 * 16 * 16 * 256);

    bconv_gemm<256, 256, 16, 16><<<dim3(32768 / 128, 256 / 64), T, 0, stream>>>(
        actA, wq4, tmp);
    postproc<true, true><<<blk((size_t)128 * 8 * 8 * 256), T, 0, stream>>>(
        tmp, 16, 16, 256, bn4s, bn4b, actB, nullptr, (size_t)128 * 8 * 8 * 256);

    bconv_gemm<256, 512, 8, 8><<<dim3(8192 / 128, 512 / 64), T, 0, stream>>>(
        actB, wq5, tmp);
    postproc<false, true><<<blk((size_t)128 * 8 * 8 * 512), T, 0, stream>>>(
        tmp, 8, 8, 512, bn5s, bn5b, actA, nullptr, (size_t)128 * 8 * 8 * 512);

    bconv_gemm<512, 512, 8, 8><<<dim3(8192 / 128, 512 / 64), T, 0, stream>>>(
        actA, wq6, tmp);
    float* denseIn = (float*)actB;
    postproc<true, false><<<blk((size_t)128 * 4 * 4 * 512), T, 0, stream>>>(
        tmp, 8, 8, 512, bn6s, bn6b, nullptr, denseIn, (size_t)128 * 4 * 4 * 512);

    dense_softmax<<<2048 / 8, T, 0, stream>>>(denseIn, dw, db, (float*)d_out);
}
